// DDSPSynthesizer_32993938767993
// MI455X (gfx1250) — matrix operand order repaired and checked by reading, not compiled, not run on hardware
//
#include <hip/hip_runtime.h>
#include <hip/hip_bf16.h>
#include <math.h>

// ---------------- problem constants ----------------
#define B_   16
#define L_   128
#define T_   62400
#define H_   32
#define V_   16
#define DZ_  128
#define DC_  16
#define DZC_ 144      // DZ + DC
#define SR_  48000.0f

typedef _Float16 v16h __attribute__((ext_vector_type(16)));
typedef _Float16 v8h  __attribute__((ext_vector_type(8)));
typedef float    v8f  __attribute__((ext_vector_type(8)));

// CDNA5 async global->LDS path (guarded: falls back cleanly if the
// toolchain doesn't declare the gfx1250 async builtins)
#if defined(__HIP_DEVICE_COMPILE__) && defined(__gfx1250__) && \
    __has_builtin(__builtin_amdgcn_global_load_async_to_lds_b32) && \
    __has_builtin(__builtin_amdgcn_s_wait_asynccnt)
#define ASYNC_LDS 1
#else
#define ASYNC_LDS 0
#endif

__device__ __forceinline__ float sigm(float x)  { return 1.0f / (1.0f + __expf(-x)); }
__device__ __forceinline__ float softp(float x) { return (x > 20.0f) ? x : __logf(1.0f + __expf(x)); }

// =====================================================================
// Weight packer: f32 W[k,Cin,Cout] (+bias) -> f16 Wp[Kpad][Coutpad] with
// zero padding (multiples of 32) + f32 Bp[Coutpad].  Row kk = dk*Cin+c,
// so flattened W is already [K][Cout].  Packed layout is plain row-major;
// a B-fragment lane reads one N column, its 16 k values Coutpad apart.
// =====================================================================
__global__ void pack_w_kernel(const float* __restrict__ W, const float* __restrict__ Bi,
                              _Float16* __restrict__ Wp, float* __restrict__ Bp,
                              int K, int Cout, int Kpad, int Coutpad)
{
    const int i = blockIdx.x * blockDim.x + threadIdx.x;
    const int total = Kpad * Coutpad;
    if (i < total) {
        const int kk = i / Coutpad, n = i - kk * Coutpad;
        float v = 0.0f;
        if (kk < K && n < Cout) v = W[(size_t)kk * Cout + n];
        Wp[i] = (_Float16)v;
    }
    if (i < Coutpad) Bp[i] = (i < Cout) ? Bi[i] : 0.0f;
}

// =====================================================================
// 1-D "SAME" conv as implicit GEMM on v_wmma_f32_16x16x32_f16.
// Templated on (CIN, KSZ): K decomposition is compile time.
// Block = 128 threads = 4 waves; each wave owns a 32(M)x32(N) tile
// = 2x2 subtiles = 4 wmma per 32-wide K chunk, A frags reused.
// M tiles are 32-aligned and Len%32==0 -> batch index constant per wave.
// CIN==1: A frags built directly from global (no LDS, no barriers).
// CIN%32==0: interior tiles use async global->LDS b32 copies.
// act: 0=none 1=relu 2=tanh
// =====================================================================
#define CWAVES 4

template<int CIN, int KSZ>
__global__ __launch_bounds__(128)
void conv1d_wmma_kernel(const _Float16* __restrict__ X,
                        const _Float16* __restrict__ Wp,   // packed [Kpad][Coutpad]
                        const float*    __restrict__ Bp,   // packed [Coutpad]
                        _Float16*       __restrict__ Y,
                        int Mtot, int Len, int Cout, int Coutpad, int act)
{
    constexpr int  K    = KSZ * CIN;
    constexpr int  KCH  = (K + 31) / 32;
    constexpr int  PAD  = KSZ / 2;
    constexpr bool FAST = (CIN % 32) == 0;    // chunk = 32 contiguous channels of one tap
    constexpr bool NOLDS = (CIN == 1);
    constexpr int  ASH_ELEMS = NOLDS ? 8 : (CWAVES * 32 * 32);

    __shared__ __align__(16) _Float16 Ash[ASH_ELEMS];     // per-wave 32M x 32K (if used)

    const int tid   = threadIdx.x;
    const int w     = tid >> 5;           // wave id (wave32)
    const int lane  = tid & 31;
    const int mtile = (blockIdx.x * CWAVES + w) * 32;
    const int nbase = blockIdx.y * 32;
    const bool wvalid = (mtile < Mtot);

    // batch / position: tile never crosses a batch seam (Len % 32 == 0)
    const int    b    = wvalid ? (mtile / Len) : 0;
    const int    l0   = mtile - b * Len;
    const size_t bLen = (size_t)b * Len;

    v8f acc00 = {}, acc01 = {}, acc10 = {}, acc11 = {};

    const int h  = lane >> 4;
    const int ml = lane & 15;

    for (int kc = 0; kc < KCH; ++kc) {
        v16h Af0, Af1;   // A fragments for M subtile 0 / 1

        if constexpr (NOLDS) {
            // ---- CIN==1: frag elements straight from global, branchless ----
            #pragma unroll
            for (int i = 0; i < 16; ++i) {
                const int kl = (i < 8) ? (8 * h + i) : (8 + 8 * h + i);
                const int kk = kc * 32 + kl;       // == tap index dk
                const int s0 = l0 + ml + kk - PAD;
                const int s1 = s0 + 16;
                const int s0c = (s0 < 0) ? 0 : ((s0 >= Len) ? (Len - 1) : s0);
                const int s1c = (s1 < 0) ? 0 : ((s1 >= Len) ? (Len - 1) : s1);
                const _Float16 x0 = X[bLen + (size_t)s0c];
                const _Float16 x1 = X[bLen + (size_t)s1c];
                const bool kv = wvalid && (kk < K);
                Af0[i] = (kv && s0 == s0c) ? x0 : (_Float16)0.0f;
                Af1[i] = (kv && s1 == s1c) ? x1 : (_Float16)0.0f;
            }
        } else {
            __syncthreads();   // protect LDS frag reads of previous chunk

            if constexpr (FAST) {
                // chunk = tap dk, channels [c0, c0+32); lane moves a b32 pair
                const int dk = (kc * 32) / CIN;
                const int c0 = (kc * 32) - dk * CIN;
                const int jp = lane & 15;            // channel pair id
                const int m0 = (lane >> 4) * 16;     // row half
                const int c  = c0 + 2 * jp;
#if ASYNC_LDS
                const bool interior =
                    wvalid && (l0 - PAD >= 0) && (l0 + 31 + (KSZ - 1) - PAD < Len);
                if (interior) {
                    #pragma unroll
                    for (int mm = 0; mm < 16; ++mm) {
                        const int m  = m0 + mm;
                        const int sl = l0 + m + dk - PAD;
                        __builtin_amdgcn_global_load_async_to_lds_b32(
                            (__attribute__((address_space(1))) int*)
                                (X + (bLen + (size_t)sl) * CIN + c),
                            (__attribute__((address_space(3))) int*)
                                (&Ash[(((w << 5) + m) << 5) + 2 * jp]),
                            0, 0);
                    }
                    __builtin_amdgcn_s_wait_asynccnt(0);
                } else
#endif
                {
                    #pragma unroll
                    for (int mm = 0; mm < 16; ++mm) {
                        const int m   = m0 + mm;
                        const int sl  = l0 + m + dk - PAD;
                        const int slc = (sl < 0) ? 0 : ((sl >= Len) ? (Len - 1) : sl);
                        unsigned v = *(const unsigned*)(X + (bLen + (size_t)slc) * CIN + c);
                        if (!(wvalid && sl == slc)) v = 0u;
                        *(unsigned*)(&Ash[(((w << 5) + m) << 5) + 2 * jp]) = v;
                    }
                }
            } else {
                // generic: lane owns one K column (compile-time magic division)
                const int kk  = kc * 32 + lane;
                const int kkc = (kk < K) ? kk : (K - 1);
                const int dk  = kkc / CIN;
                const int c   = kkc - dk * CIN;
                const bool kok = (kk < K) && wvalid;
                #pragma unroll 4
                for (int m = 0; m < 32; ++m) {
                    const int sl  = l0 + m + dk - PAD;
                    const int slc = (sl < 0) ? 0 : ((sl >= Len) ? (Len - 1) : sl);
                    _Float16 v = X[(bLen + (size_t)slc) * CIN + c];
                    if (!(kok && sl == slc)) v = (_Float16)0.0f;
                    Ash[(((w << 5) + m) << 5) + lane] = v;
                }
            }
            __syncthreads();

            // A fragments from LDS (two aligned b128 reads per subtile)
            {
                const v8h* ar = (const v8h*)(&Ash[((w << 5) + ml) << 5]);
                const v8h a0 = ar[h], a1 = ar[2 + h];
                #pragma unroll
                for (int i = 0; i < 8; ++i) { Af0[i] = a0[i]; Af0[i + 8] = a1[i]; }
            }
            {
                const v8h* ar = (const v8h*)(&Ash[((w << 5) + 16 + ml) << 5]);
                const v8h a0 = ar[h], a1 = ar[2 + h];
                #pragma unroll
                for (int i = 0; i < 8; ++i) { Af1[i] = a0[i]; Af1[i + 8] = a1[i]; }
            }
        }

        // ---------------- B fragments: direct packed global loads ----------------
        const _Float16* wrow = Wp + (size_t)(kc * 32 + lane) * Coutpad + nbase;
        if (kc + 1 < KCH && lane == 0)
            __builtin_prefetch(wrow + (size_t)32 * Coutpad, 0, 1);   // global_prefetch_b8
        const _Float16* wcol = Wp + (size_t)(kc * 32) * Coutpad + nbase + ml;
        v16h Bf0, Bf1;
        #pragma unroll
        for (int i = 0; i < 16; ++i) {
            const int kl = (i < 8) ? (8 * h + i) : (16 + 8 * h + (i - 8));
            Bf0[i] = wcol[(size_t)kl * Coutpad];
            Bf1[i] = wcol[(size_t)kl * Coutpad + 16];
        }

        // ---------------- 4 WMMA, A frags reused across N ----------------
        acc00 = __builtin_amdgcn_wmma_f32_16x16x32_f16(false, Af0, false, Bf0, (short)0, acc00, false, false);
        acc01 = __builtin_amdgcn_wmma_f32_16x16x32_f16(false, Af0, false, Bf1, (short)0, acc01, false, false);
        acc10 = __builtin_amdgcn_wmma_f32_16x16x32_f16(false, Af1, false, Bf0, (short)0, acc10, false, false);
        acc11 = __builtin_amdgcn_wmma_f32_16x16x32_f16(false, Af1, false, Bf1, (short)0, acc11, false, false);
    }

    // ---------------- epilogue: bias + act, store f16 ----------------
    // C/D layout: VGPR r, lane -> M = r + 8*(lane>>4), N = lane&15
    if (!wvalid) return;
    const int nl = lane & 15;
    #pragma unroll
    for (int ns = 0; ns < 2; ++ns) {
        const int n = nbase + ns * 16 + nl;
        if (n >= Cout) continue;
        const float bia = Bp[n];
        #pragma unroll
        for (int ms = 0; ms < 2; ++ms) {
            const v8f& a = (ms == 0) ? (ns == 0 ? acc00 : acc01)
                                     : (ns == 0 ? acc10 : acc11);
            #pragma unroll
            for (int r = 0; r < 8; ++r) {
                const int row = mtile + ms * 16 + r + 8 * h;
                if (row < Mtot) {
                    float v = a[r] + bia;
                    if      (act == 1) v = fmaxf(v, 0.0f);
                    else if (act == 2) v = tanhf(v);
                    Y[(size_t)row * Cout + n] = (_Float16)v;
                }
            }
        }
    }
}

// host-side template dispatch over the layer shapes that actually occur
static void conv_dispatch(const _Float16* X, const _Float16* Wp, const float* Bp,
                          _Float16* Y, int Mtot, int Len, int Cin, int Cout,
                          int Coutpad, int ks, int act, hipStream_t stream)
{
    dim3 grid((Mtot + 127) / 128, (Cout + 31) / 32);
#define CONV_CASE(CI, KS)                                                        \
    if (Cin == CI && ks == KS) {                                                 \
        conv1d_wmma_kernel<CI, KS><<<grid, 128, 0, stream>>>(                    \
            X, Wp, Bp, Y, Mtot, Len, Cout, Coutpad, act);                        \
        return;                                                                  \
    }
    CONV_CASE(144, 9)  CONV_CASE(144, 7)
    CONV_CASE(256, 7)  CONV_CASE(256, 5)
    CONV_CASE(128, 7)  CONV_CASE(128, 5)
    CONV_CASE(64, 11)  CONV_CASE(64, 5)   CONV_CASE(64, 3)
    CONV_CASE(32, 7)   CONV_CASE(32, 3)
    CONV_CASE(16, 5)
    CONV_CASE(1, 15)
#undef CONV_CASE
}

// =====================================================================
// Small dense layers (M = 16 rows only) -- VALU is fine here.
// act: 0=none 1=relu 3=sigmoid
// =====================================================================
__global__ void dense_kernel(const float* __restrict__ X, const float* __restrict__ W,
                             const float* __restrict__ Bi, float* __restrict__ Y,
                             int Brows, int In, int Out, int act)
{
    const int i = blockIdx.x * blockDim.x + threadIdx.x;
    if (i >= Brows * Out) return;
    const int b = i / Out, o = i - b * Out;
    float s = Bi[o];
    for (int k = 0; k < In; ++k) s += X[b * In + k] * W[k * Out + o];
    if      (act == 1) s = fmaxf(s, 0.0f);
    else if (act == 3) s = sigm(s);
    Y[i] = s;
}

// ---------- build zc (f16) = concat(z, broadcast cond) ----------
__global__ void prep_zc_kernel(const float* __restrict__ z, const float* __restrict__ cond,
                               _Float16* __restrict__ zc)
{
    const int i = blockIdx.x * blockDim.x + threadIdx.x;
    if (i >= B_ * L_ * DZC_) return;
    const int c  = i % DZC_;
    const int bl = i / DZC_;
    const int b  = bl / L_;
    const float v = (c < DZ_) ? z[(size_t)bl * DZ_ + c] : cond[b * DC_ + (c - DZ_)];
    zc[i] = (_Float16)v;
}

// ---------- comb (f32) = concat(mean_L z, cond) ----------
__global__ void comb_kernel(const float* __restrict__ z, const float* __restrict__ cond,
                            float* __restrict__ comb)
{
    const int i = blockIdx.x * blockDim.x + threadIdx.x;
    if (i >= B_ * DZC_) return;
    const int b = i / DZC_, c = i % DZC_;
    float v;
    if (c < DZ_) {
        float s = 0.0f;
        for (int l = 0; l < L_; ++l) s += z[((size_t)b * L_ + l) * DZ_ + c];
        v = s * (1.0f / L_);
    } else {
        v = cond[b * DC_ + (c - DZ_)];
    }
    comb[i] = v;
}

__global__ void f32_to_f16_kernel(const float* __restrict__ x, _Float16* __restrict__ y, int n)
{
    const int i = blockIdx.x * blockDim.x + threadIdx.x;
    if (i < n) y[i] = (_Float16)x[i];
}

__device__ __forceinline__ void resize_coeff(int t, int& i0, int& i1, float& wt)
{
    float src = (t + 0.5f) * ((float)L_ / (float)T_) - 0.5f;
    src = fminf(fmaxf(src, 0.0f), (float)(L_ - 1));
    i0 = (int)src;
    i1 = (i0 + 1 < L_) ? i0 + 1 : L_ - 1;
    wt = src - (float)i0;
}

// ---------- additive harmonic synthesis (fused resize+sigmoid+sin) ----------
__global__ void harmonic_kernel(const _Float16* __restrict__ ampsL,  // [B,L,H]
                                const float* __restrict__ fs,        // [B,H]
                                const float* __restrict__ f0v,       // [B]
                                const float* __restrict__ phases,    // [B,H]
                                float* __restrict__ harm)            // [B,T]
{
    const int i = blockIdx.x * blockDim.x + threadIdx.x;
    if (i >= B_ * T_) return;
    const int b = i / T_, t = i - b * T_;
    int i0, i1; float wt;
    resize_coeff(t, i0, i1, wt);
    const _Float16* r0 = ampsL + ((size_t)b * L_ + i0) * H_;
    const _Float16* r1 = ampsL + ((size_t)b * L_ + i1) * H_;
    const float f    = f0v[b];
    const float base = 6.28318530718f * f * ((float)t / SR_);
    float acc = 0.0f;
    #pragma unroll 4
    for (int k = 0; k < H_; ++k) {
        const float fr = f * (float)(k + 1);
        if (fr < 0.5f * SR_) {
            float a = (float)r0[k] * (1.0f - wt) + (float)r1[k] * wt + 0.5f * fs[b * H_ + k];
            a = sigm(a);
            acc += a * __sinf(base * (float)(k + 1) + phases[b * H_ + k]);
        }
    }
    harm[i] = acc;
}

// ---------- fused unison + envelope + noise mix -> output ----------
__global__ void final_kernel(const float* __restrict__ harm,      // [B,T]
                             const _Float16* __restrict__ gainsL, // [B,L,V]
                             const float* __restrict__ p4,        // [B,4]
                             const _Float16* __restrict__ envL,   // [B,L]
                             const _Float16* __restrict__ modL,   // [B,L]
                             const _Float16* __restrict__ nenvL,  // [B,L]
                             const _Float16* __restrict__ filt,   // [B,T]
                             const float* __restrict__ amount,    // [B] (already sigmoid)
                             float* __restrict__ out)             // [B,T]
{
    const int i = blockIdx.x * blockDim.x + threadIdx.x;
    if (i >= B_ * T_) return;
    const int b = i / T_, t = i - b * T_;
    int i0, i1; float wt;
    resize_coeff(t, i0, i1, wt);

    const float nv     = 1.0f + 14.0f * sigm(p4[b * 4 + 0]);
    const float spread = sigm(p4[b * 4 + 2]);
    const float pmd    = sigm(p4[b * 4 + 3]) * 0.5f;
    const float tsec   = (float)t / SR_;

    const _Float16* g0 = gainsL + ((size_t)b * L_ + i0) * V_;
    const _Float16* g1 = gainsL + ((size_t)b * L_ + i1) * V_;

    float uni = 0.0f, gsum = 0.0f, vsum = 0.0f;
    #pragma unroll
    for (int v = 0; v < V_; ++v) {
        const float pos = ((float)v - 7.5f) / 16.0f;
        const int   sft = (int)(pos * 20.0f);          // trunc toward 0 == Python int()
        int ts = t - sft;
        if (ts < 0) ts += T_; else if (ts >= T_) ts -= T_;  // roll
        const float hs = harm[(size_t)b * T_ + ts];

        const float g  = softp((float)g0[v] * (1.0f - wt) + (float)g1[v] * wt);
        const float vm = sigm((nv - (float)v) * 2.0f);
        vsum += vm;
        gsum += g * vm;

        const float lfo = pmd * __sinf(6.28318530718f * (3.0f + 0.3f * (float)v) * tsec);
        const float pan = 1.0f - fabsf(pos) * spread * 0.5f;
        uni += hs * (1.0f + 0.2f * lfo) * g * pan;
    }
    const float norm = sqrtf(vsum + 1e-6f);
    uni = gsum / (norm + 1e-6f) * uni;

    const float ev = (float)envL[b * L_ + i0] * (1.0f - wt) + (float)envL[b * L_ + i1] * wt;
    const float mv = (float)modL[b * L_ + i0] * (1.0f - wt) + (float)modL[b * L_ + i1] * wt;
    const float envelope = sigm(ev + 0.3f * mv);

    const float nev  = (float)nenvL[b * L_ + i0] * (1.0f - wt) + (float)nenvL[b * L_ + i1] * wt;
    const float nenv = sigm(nev);
    const float ns   = (float)filt[i] * nenv * (amount[b] * 0.02f);

    out[i] = uni * envelope + ns;
}

// =====================================================================
// Host side
// =====================================================================
extern "C" void kernel_launch(void* const* d_in, const int* in_sizes, int n_in,
                              void* d_out, int out_size, void* d_ws, size_t ws_size,
                              hipStream_t stream)
{
    (void)in_sizes; (void)n_in; (void)out_size; (void)ws_size;

    // ---- input tensor order (setup_inputs dict order, depth-first params) ----
    const float* z      = (const float*)d_in[0];
    const float* cond   = (const float*)d_in[1];
    const float* f0     = (const float*)d_in[2];
    const float* noise  = (const float*)d_in[3];
    const float* phases = (const float*)d_in[4];
    // param leaf indices:
    // env.envelope w/b: 5..14   env.modulation: 15..20
    // harm.net: 21..30          harm.freq_shaping: 31..36
    // noise.env: 37..42         noise.filter: 43..50   noise.amount: 51..56
    // unison.pred: 57..64       unison.gains: 65..70
    float* out = (float*)d_out;

    // ---- workspace carve-out (deterministic sequential allocation) ----
    char*  ws  = (char*)d_ws;
    size_t off = 0;
    auto alloc = [&](size_t bytes) -> void* {
        off = (off + 255) & ~(size_t)255;
        void* p = ws + off;
        off += bytes;
        return p;
    };
    _Float16* zc      = (_Float16*)alloc((size_t)B_ * L_ * DZC_ * 2);
    float*    comb    = (float*)   alloc((size_t)B_ * DZC_ * 4);
    _Float16* pingL   = (_Float16*)alloc((size_t)B_ * L_ * 256 * 2);
    _Float16* pongL   = (_Float16*)alloc((size_t)B_ * L_ * 256 * 2);
    _Float16* ampsL   = (_Float16*)alloc((size_t)B_ * L_ * H_ * 2);
    _Float16* envL    = (_Float16*)alloc((size_t)B_ * L_ * 2);
    _Float16* modL    = (_Float16*)alloc((size_t)B_ * L_ * 2);
    _Float16* nenvL   = (_Float16*)alloc((size_t)B_ * L_ * 2);
    _Float16* gainsL  = (_Float16*)alloc((size_t)B_ * L_ * V_ * 2);
    float*    fsb     = (float*)   alloc((size_t)B_ * H_ * 4);
    float*    amount  = (float*)   alloc((size_t)B_ * 4);
    float*    p4      = (float*)   alloc((size_t)B_ * 4 * 4);
    float*    dbufA   = (float*)   alloc((size_t)B_ * 256 * 4);
    float*    dbufB   = (float*)   alloc((size_t)B_ * 256 * 4);
    _Float16* noise16 = (_Float16*)alloc((size_t)B_ * T_ * 2);
    _Float16* nf1     = (_Float16*)alloc((size_t)B_ * T_ * 64 * 2);  // reused for nf3
    _Float16* nf2     = (_Float16*)alloc((size_t)B_ * T_ * 32 * 2);  // reused for filt
    float*    harm    = (float*)   alloc((size_t)B_ * T_ * 4);
    _Float16* nf3  = nf1;   // layer-3 output overwrites dead layer-1 buffer
    _Float16* filt = nf2;   // layer-4 output overwrites dead layer-2 buffer

    // ---- conv launcher: pack weights (f32->f16, row-major [Kpad][Coutpad], zero pad), run ----
    auto conv = [&](const _Float16* X, int wi, _Float16* Y,
                    int Mtot, int Len, int Cin, int Cout, int ks, int act) {
        const int K       = ks * Cin;
        const int Kpad    = (K + 31) & ~31;
        const int Coutpad = (Cout + 31) & ~31;
        _Float16* Wp = (_Float16*)alloc((size_t)Kpad * Coutpad * 2);
        float*    Bp = (float*)alloc((size_t)Coutpad * 4);
        const int npk = Kpad * Coutpad;
        pack_w_kernel<<<(npk + 255) / 256, 256, 0, stream>>>(
            (const float*)d_in[wi], (const float*)d_in[wi + 1], Wp, Bp,
            K, Cout, Kpad, Coutpad);
        conv_dispatch(X, Wp, Bp, Y, Mtot, Len, Cin, Cout, Coutpad, ks, act, stream);
    };
    auto dense = [&](const float* X, int wi, float* Y, int In, int Out, int act) {
        int n = B_ * Out;
        dense_kernel<<<(n + 127) / 128, 128, 0, stream>>>(
            X, (const float*)d_in[wi], (const float*)d_in[wi + 1], Y, B_, In, Out, act);
    };

    const int ML = B_ * L_;     // 2048
    const int MT = B_ * T_;     // 998400

    // ---- prep ----
    {
        int n = B_ * L_ * DZC_;
        prep_zc_kernel<<<(n + 255) / 256, 256, 0, stream>>>(z, cond, zc);
        int m = B_ * DZC_;
        comb_kernel<<<(m + 127) / 128, 128, 0, stream>>>(z, cond, comb);
        f32_to_f16_kernel<<<(MT + 255) / 256, 256, 0, stream>>>(noise, noise16, MT);
    }

    // ---- harm.net: 144->256(k9)->256(k7)->128(k5)->64(k5)->32(k3) ----
    conv(zc,    21, pingL, ML, L_, DZC_, 256, 9, 1);
    conv(pingL, 23, pongL, ML, L_, 256,  256, 7, 1);
    conv(pongL, 25, pingL, ML, L_, 256,  128, 5, 1);
    conv(pingL, 27, pongL, ML, L_, 128,  64,  5, 1);
    conv(pongL, 29, ampsL, ML, L_, 64,   H_,  3, 0);

    // ---- env.envelope: 144->128(k9)->128(k7)->64(k5)->32(k5)->1(k3) ----
    conv(zc,    5,  pingL, ML, L_, DZC_, 128, 9, 1);
    conv(pingL, 7,  pongL, ML, L_, 128,  128, 7, 1);
    conv(pongL, 9,  pingL, ML, L_, 128,  64,  5, 1);
    conv(pingL, 11, pongL, ML, L_, 64,   32,  5, 1);
    conv(pongL, 13, envL,  ML, L_, 32,   1,   3, 0);

    // ---- env.modulation: 144->64(k7)->32(k5)->1(k3) ----
    conv(zc,    15, pingL, ML, L_, DZC_, 64, 7, 1);
    conv(pingL, 17, pongL, ML, L_, 64,   32, 5, 1);
    conv(pongL, 19, modL,  ML, L_, 32,   1,  3, 0);

    // ---- noise.env: 144->64(k7)->32(k5)->1(k3) ----
    conv(zc,    37, pingL, ML, L_, DZC_, 64, 7, 1);
    conv(pingL, 39, pongL, ML, L_, 64,   32, 5, 1);
    conv(pongL, 41, nenvL, ML, L_, 32,   1,  3, 0);

    // ---- unison.gains: 144->128(k7)->64(k5)->16(k3) ----
    conv(zc,    65, pingL,  ML, L_, DZC_, 128, 7, 1);
    conv(pingL, 67, pongL,  ML, L_, 128,  64,  5, 1);
    conv(pongL, 69, gainsL, ML, L_, 64,   V_,  3, 0);

    // ---- dense nets over comb [16,144] ----
    dense(comb,  31, dbufA, DZC_, 128, 1);   // freq_shaping
    dense(dbufA, 33, dbufB, 128,  64,  1);
    dense(dbufB, 35, fsb,   64,   H_,  0);

    dense(comb,  51, dbufA, DZC_, 64, 1);    // amount (final sigmoid; *0.02 in final kernel)
    dense(dbufA, 53, dbufB, 64,   32, 1);
    dense(dbufB, 55, amount, 32,  1,  3);

    dense(comb,  57, dbufA, DZC_, 256, 1);   // unison.pred
    dense(dbufA, 59, dbufB, 256,  128, 1);
    dense(dbufB, 61, dbufA, 128,  64,  1);
    dense(dbufA, 63, p4,    64,   4,   0);

    // ---- noise.filter over full T: 1->64(k15)->32(k11)->16(k7)->1(k5,tanh) ----
    conv(noise16, 43, nf1,  MT, T_, 1,  64, 15, 1);
    conv(nf1,     45, nf2,  MT, T_, 64, 32, 11, 1);
    conv(nf2,     47, nf3,  MT, T_, 32, 16, 7,  1);
    conv(nf3,     49, filt, MT, T_, 16, 1,  5,  2);

    // ---- additive synthesis ----
    harmonic_kernel<<<(MT + 255) / 256, 256, 0, stream>>>(ampsL, fsb, f0, phases, harm);

    // ---- unison + envelope + noise mix ----
    final_kernel<<<(MT + 255) / 256, 256, 0, stream>>>(
        harm, gainsL, p4, envL, modL, nenvL, filt, amount, out);
}
